// DoubleLayeredGraphEncoderCAT_58334245814643
// MI455X (gfx1250) — compile-verified, matrix operand order checked by reading, not run on hardware
//
#include <hip/hip_runtime.h>

typedef __attribute__((ext_vector_type(16))) _Float16 v16h;
typedef __attribute__((ext_vector_type(8)))  float    v8f;

#define IN_CH 128
#define HID   64
// packed B fragments: 4 ntiles x 4 kchunks x 32 lanes x 16 halves = 16 KB
#define NFRAG (4 * 4 * 32 * 16)

__device__ __forceinline__ v8f wmma_f16(v16h a, v16h b, v8f c)
{
    return __builtin_amdgcn_wmma_f32_16x16x32_f16(false, a, false, b,
                                                  (short)0, c, false, false);
}

// Stage W (128x64 f32, row-major) into LDS pre-swizzled into WMMA B-operand
// fragment order, so each (ntile, kchunk, lane) reads its 16 halves as one
// contiguous 32-byte vector (2x ds_load_b128).
// Fragment layout per ISA: lane 0-15 = col N=lane, halves 0..7 -> K 0..7,
// halves 8..15 -> K 16..23; lanes 16-31 same cols, K+8 / K+24.
__device__ __forceinline__ void stage_Bfrag(const float* __restrict__ W,
                                            _Float16* sB)
{
    for (int i = threadIdx.x; i < NFRAG; i += blockDim.x) {
        const int j     = i & 15;
        const int lane  = (i >> 4) & 31;
        const int kci   = (i >> 9) & 3;
        const int ntile = i >> 11;
        const int koff  = (lane >> 4) << 3;
        const int col   = (ntile << 4) + (lane & 15);
        const int k     = (kci << 5) + koff + (j & 7) + ((j >> 3) << 4);
        sB[i] = (_Float16)W[k * HID + col];
    }
}

// A fragment (16x32 f16) from a row-major f32 row: halves 0..7 = K off..off+7,
// halves 8..15 = K off+16..off+23  (off = kchunk*32 + lane-group offset).
__device__ __forceinline__ v16h load_a16(const float* __restrict__ arow, int off)
{
    float af[16];
    reinterpret_cast<float4*>(af)[0] = *reinterpret_cast<const float4*>(arow + off);
    reinterpret_cast<float4*>(af)[1] = *reinterpret_cast<const float4*>(arow + off + 4);
    reinterpret_cast<float4*>(af)[2] = *reinterpret_cast<const float4*>(arow + off + 16);
    reinterpret_cast<float4*>(af)[3] = *reinterpret_cast<const float4*>(arow + off + 20);
    v16h a;
    #pragma unroll
    for (int j = 0; j < 16; ++j) a[j] = (_Float16)af[j];
    return a;
}

// ---------------------------------------------------------------------------
// Kernel 1: h = x @ W_conv.  One wave owns 1 ntile x 4 mtiles; the four B
// fragments stay register-resident across the 4 output tiles.
// ---------------------------------------------------------------------------
__global__ void __launch_bounds__(256)
gemm1_wmma(const float* __restrict__ x,
           const float* __restrict__ Wc,
           float* __restrict__ h,
           int n_nodes)
{
    __shared__ __align__(32) _Float16 sB[NFRAG];
    stage_Bfrag(Wc, sB);
    __syncthreads();

    const int wave = threadIdx.x >> 5;
    const int lane = threadIdx.x & 31;
    const int mtiles  = n_nodes >> 4;            // 3125 (M exact multiple of 16)
    const int mgroups = (mtiles + 3) >> 2;
    const int g = blockIdx.x * (blockDim.x >> 5) + wave;
    if (g >= mgroups * 4) return;                // wave-uniform, EXEC stays full
    const int ntile = g & 3;
    const int mg    = g >> 2;

    const v16h* bp = reinterpret_cast<const v16h*>(sB) + (ntile << 7) + lane;
    const v16h b0 = bp[0], b1 = bp[32], b2 = bp[64], b3 = bp[96];

    const int koff = (lane >> 4) << 3;           // 0 or 8
    const int col  = (ntile << 4) + (lane & 15);

    for (int t = 0; t < 4; ++t) {
        const int mtile = (mg << 2) + t;
        if (mtile >= mtiles) return;             // uniform
        const int row = (mtile << 4) + (lane & 15);
        const float* arow = x + (long)row * IN_CH;
        __builtin_prefetch(arow + 256, 0, 1);    // two rows ahead (global_prefetch_b8)

        v8f c = {};
        c = wmma_f16(load_a16(arow,      koff), b0, c);
        c = wmma_f16(load_a16(arow, 32 + koff), b1, c);
        c = wmma_f16(load_a16(arow, 64 + koff), b2, c);
        c = wmma_f16(load_a16(arow, 96 + koff), b3, c);

        const int rbase = (mtile << 4) + ((lane >> 4) << 3);
        #pragma unroll
        for (int r = 0; r < 8; ++r)
            h[(long)(rbase + r) * HID + col] = c[r];
    }
}

// ---------------------------------------------------------------------------
// Kernel 2: zero agg (re-zeroed every call for graph-replay determinism)
// ---------------------------------------------------------------------------
__global__ void zero_f32(float4* __restrict__ p, int n4)
{
    int i = blockIdx.x * blockDim.x + threadIdx.x;
    for (; i < n4; i += gridDim.x * blockDim.x)
        p[i] = make_float4(0.f, 0.f, 0.f, 0.f);
}

// ---------------------------------------------------------------------------
// Kernel 3: edge scatter.  One wave per edge, 2 channels per lane.
// agg[dst] += h[src] * w  via global_atomic_add_f32 (L2-resident region).
// ---------------------------------------------------------------------------
__global__ void __launch_bounds__(256)
scatter_edges(const float* __restrict__ h,
              const int*   __restrict__ ei,
              const float* __restrict__ ew,
              float*       __restrict__ agg,
              int E)
{
    const int t    = blockIdx.x * blockDim.x + threadIdx.x;
    const int e    = t >> 5;
    const int lane = t & 31;
    if (e >= E) return;

    const int   s = ei[e];
    const int   d = ei[E + e];
    const float w = ew[e];

    float2 hv = reinterpret_cast<const float2*>(h + (long)s * HID)[lane];
    float* ap = agg + (long)d * HID + (lane << 1);
    atomicAdd(ap,     hv.x * w);
    atomicAdd(ap + 1, hv.y * w);
}

// ---------------------------------------------------------------------------
// Kernel 4: in-place agg = lrelu(agg + b_conv)  (one float4 per thread)
// ---------------------------------------------------------------------------
__global__ void __launch_bounds__(256)
bias_lrelu(float* __restrict__ agg,
           const float* __restrict__ b_conv,
           const float* __restrict__ alpha,
           int total4)
{
    const int i = blockIdx.x * blockDim.x + threadIdx.x;
    if (i >= total4) return;
    const int c4 = (i & 15) << 2;              // channel base (HID/4 == 16)
    float4 v  = reinterpret_cast<float4*>(agg)[i];
    float4 bb = *reinterpret_cast<const float4*>(b_conv + c4);
    float4 aa = *reinterpret_cast<const float4*>(alpha + c4);
    v.x += bb.x; v.y += bb.y; v.z += bb.z; v.w += bb.w;
    v.x = v.x > 0.f ? v.x : aa.x * v.x;
    v.y = v.y > 0.f ? v.y : aa.y * v.y;
    v.z = v.z > 0.f ? v.z : aa.z * v.z;
    v.w = v.w > 0.f ? v.w : aa.w * v.w;
    reinterpret_cast<float4*>(agg)[i] = v;
}

// ---------------------------------------------------------------------------
// Kernel 5: out = relu( [act(:25000) | act(25000:)] @ W_cat + b_cat )
// Concat handled as a per-K-chunk row offset (part = kchunk>=2).
// M = 25000 -> last tile partially valid: clamp loads, predicate stores.
// ---------------------------------------------------------------------------
__global__ void __launch_bounds__(256)
gemm2_wmma(const float* __restrict__ act,
           const float* __restrict__ Wcat,
           const float* __restrict__ b_cat,
           float* __restrict__ out,
           int n_half)
{
    __shared__ __align__(32) _Float16 sB[NFRAG];
    stage_Bfrag(Wcat, sB);
    __syncthreads();

    const int wave = threadIdx.x >> 5;
    const int lane = threadIdx.x & 31;
    const int mtiles  = (n_half + 15) >> 4;      // 1563
    const int mgroups = (mtiles + 3) >> 2;
    const int g = blockIdx.x * (blockDim.x >> 5) + wave;
    if (g >= mgroups * 4) return;                // wave-uniform
    const int ntile = g & 3;
    const int mg    = g >> 2;

    const v16h* bp = reinterpret_cast<const v16h*>(sB) + (ntile << 7) + lane;
    const v16h b0 = bp[0], b1 = bp[32], b2 = bp[64], b3 = bp[96];

    const int koff = (lane >> 4) << 3;
    const int col  = (ntile << 4) + (lane & 15);

    for (int t = 0; t < 4; ++t) {
        const int mtile = (mg << 2) + t;
        if (mtile >= mtiles) return;             // uniform
        const int row  = (mtile << 4) + (lane & 15);
        const int rowc = row < n_half ? row : n_half - 1;    // clamp OOB loads
        const float* a0 = act + (long)rowc * HID;            // k 0..63
        const float* a1 = act + (long)(rowc + n_half) * HID; // k 64..127

        v8f c = {};
        c = wmma_f16(load_a16(a0,      koff), b0, c);
        c = wmma_f16(load_a16(a0, 32 + koff), b1, c);
        c = wmma_f16(load_a16(a1,      koff), b2, c);
        c = wmma_f16(load_a16(a1, 32 + koff), b3, c);

        const float bb = b_cat[col];
        const int rbase = (mtile << 4) + ((lane >> 4) << 3);
        #pragma unroll
        for (int r = 0; r < 8; ++r) {
            const int rr = rbase + r;
            if (rr < n_half) {
                float v = c[r] + bb;
                out[(long)rr * HID + col] = v > 0.f ? v : 0.f;
            }
        }
    }
}

// ---------------------------------------------------------------------------
extern "C" void kernel_launch(void* const* d_in, const int* in_sizes, int n_in,
                              void* d_out, int out_size, void* d_ws, size_t ws_size,
                              hipStream_t stream)
{
    const float* x   = (const float*)d_in[0];   // (n_nodes, 128)
    const int*   ei  = (const int*)  d_in[1];   // (2, E)
    const float* ew  = (const float*)d_in[2];   // (E,)
    const float* Wc  = (const float*)d_in[3];   // (128, 64)
    const float* bc  = (const float*)d_in[4];   // (64,)
    const float* al  = (const float*)d_in[5];   // (64,)
    const float* Wk  = (const float*)d_in[6];   // (128, 64)
    const float* bk  = (const float*)d_in[7];   // (64,)

    const int n_nodes = in_sizes[0] / IN_CH;    // 50000
    const int E       = in_sizes[2];            // 800000
    const int n_half  = n_nodes / 2;            // 25000

    float* h   = (float*)d_ws;                  // n_nodes*64 f32
    float* agg = h + (size_t)n_nodes * HID;     // n_nodes*64 f32
    float* out = (float*)d_out;

    // 1) h = x @ W_conv
    {
        const int mgroups = ((n_nodes >> 4) + 3) >> 2;
        const int waves   = mgroups * 4;
        gemm1_wmma<<<(waves + 7) / 8, 256, 0, stream>>>(x, Wc, h, n_nodes);
    }

    // 2) agg = 0
    const int n4 = (n_nodes * HID) >> 2;
    zero_f32<<<(n4 + 255) / 256, 256, 0, stream>>>((float4*)agg, n4);

    // 3) agg[dst] += h[src] * w
    scatter_edges<<<(E + 7) / 8, 256, 0, stream>>>(h, ei, ew, agg, E);

    // 4) agg = lrelu(agg + b_conv)   (in place; agg is rebuilt every call)
    bias_lrelu<<<(n4 + 255) / 256, 256, 0, stream>>>(agg, bc, al, n4);

    // 5) fused concat GEMM + bias + ReLU
    {
        const int mgroups = (((n_half + 15) >> 4) + 3) >> 2;
        const int waves   = mgroups * 4;
        gemm2_wmma<<<(waves + 7) / 8, 256, 0, stream>>>(agg, Wk, bk, out, n_half);
    }
}
